// MambaEncoder_16252156248264
// MI455X (gfx1250) — compile-verified, matrix operand order checked by reading, not run on hardware
//
#include <hip/hip_runtime.h>
#include <hip/hip_bf16.h>

typedef __attribute__((ext_vector_type(16))) _Float16 v16h;
typedef __attribute__((ext_vector_type(8)))  _Float16 v8h;
typedef __attribute__((ext_vector_type(8)))  float    v8f;

#define BATCH   4
#define SEQ     2048
#define D_MODEL 128
#define D_FF    256
#define D_INNER 256
#define D_STATE 16
#define DT_RANK 8
#define MROWS   (BATCH * SEQ)      // 8192
#define DBC_LD  48                 // padded leading dim for 40-wide dbc

// ---------------------------------------------------------------------------
// f16 fragment loader (CDNA5 ISA 7.12.2 16-bit A/B layout).
// Per lane, 16 halves = two contiguous 8-half (16 B) runs:
//   k = half*8 + {0..7}  and  k = 16 + half*8 + {0..7}
// -> exactly 2x global_load_b128, zero converts.
// ---------------------------------------------------------------------------
__device__ __forceinline__ v16h load_frag_h(const _Float16* __restrict__ p, int half) {
    v8h lo = *(const v8h*)(p + (half << 3));
    v8h hi = *(const v8h*)(p + 16 + (half << 3));
    v16h f;
#pragma unroll
    for (int e = 0; e < 8; ++e) { f[e] = lo[e]; f[8 + e] = hi[e]; }
    return f;
}

// ---------------------------------------------------------------------------
// WMMA f16 GEMM:  C[M,N] = A[M,K] * W[N,K]^T  (+bias, +relu per EPI)
// One wave computes a 16 x (16*NTILE) strip; A fragment amortized over NTILE
// B fragments.  All NTILE+1 fragments are loaded into distinct registers
// before the WMMAs so loads overlap (single wait per k-step).
// Out-of-range columns are CLAMPED on load, guarded only at the store
// (no EXEC divergence in the loop; WMMA requires EXEC all-ones).
// EPI: 0 = none, 1 = bias+relu, 2 = bias
// OST: 0 = f32 store, 1 = f16 store, 2 = both
// flipL > 0 : remap A row l -> (flipL-1-l) within each length-flipL segment
// ---------------------------------------------------------------------------
template <int EPI, int NTILE, int OST>
__global__ void wmma_gemm(const _Float16* __restrict__ A, const _Float16* __restrict__ W,
                          const float* __restrict__ bias, float* __restrict__ C,
                          _Float16* __restrict__ Ch,
                          int Mr, int Nr, int Kr, int lda, int ldw, int ldc, int flipL) {
    int wave = (blockIdx.x * blockDim.x + threadIdx.x) >> 5;
    int lane = threadIdx.x & 31;
    int ngrp = (Nr + 16 * NTILE - 1) / (16 * NTILE);
    int mt = wave / ngrp;
    int ng = wave - mt * ngrp;
    int m0 = mt << 4;
    int n0 = ng * (16 * NTILE);
    if (m0 >= Mr) return;                   // uniform per wave

    int half = lane >> 4;
    int arow = m0 + (lane & 15);
    if (flipL > 0) {
        int l = arow % flipL;
        arow += flipL - 1 - 2 * l;          // flip within sequence
    }
    const _Float16* __restrict__ Arow = A + (size_t)arow * lda;

    const _Float16* Wrow[NTILE];
    int wcol[NTILE];
#pragma unroll
    for (int t = 0; t < NTILE; ++t) {
        int c   = n0 + 16 * t + (lane & 15);
        wcol[t] = c;
        int cc  = (c < Nr) ? c : (Nr - 1);  // clamp: loads stay in-bounds
        Wrow[t] = W + (size_t)cc * ldw;
    }

    __builtin_prefetch(Arow, 0, 1);         // global_prefetch_b8
    __builtin_prefetch(Wrow[0], 0, 1);

    v8f acc[NTILE];
#pragma unroll
    for (int t = 0; t < NTILE; ++t)
#pragma unroll
        for (int e = 0; e < 8; ++e) acc[t][e] = 0.0f;

    for (int k0 = 0; k0 < Kr; k0 += 32) {
        v16h af = load_frag_h(Arow + k0, half);
        v16h bf[NTILE];
#pragma unroll
        for (int t = 0; t < NTILE; ++t)
            bf[t] = load_frag_h(Wrow[t] + k0, half);
#pragma unroll
        for (int t = 0; t < NTILE; ++t)
            acc[t] = __builtin_amdgcn_wmma_f32_16x16x32_f16(
                /*neg_a=*/false, af, /*neg_b=*/false, bf[t],
                /*c_mod=*/(short)0, acc[t], /*reuse_a=*/false, /*reuse_b=*/false);
    }

#pragma unroll
    for (int t = 0; t < NTILE; ++t) {
        if (wcol[t] < Nr) {
            float bv = (EPI > 0) ? bias[wcol[t]] : 0.0f;
#pragma unroll
            for (int v = 0; v < 8; ++v) {
                int   row = m0 + v + (half ? 8 : 0);
                float val = acc[t][v] + bv;
                if (EPI == 1) val = fmaxf(val, 0.0f);
                if (OST == 0 || OST == 2) C[(size_t)row * ldc + wcol[t]]  = val;
                if (OST == 1 || OST == 2) Ch[(size_t)row * ldc + wcol[t]] = (_Float16)val;
            }
        }
    }
}

// ---------------------------------------------------------------------------
// One-shot f32 -> f16 conversion (x and the 8 weight matrices).
// ---------------------------------------------------------------------------
__global__ void cvt_f16_kernel(const float* __restrict__ src, _Float16* __restrict__ dst, int n) {
    int i = blockIdx.x * blockDim.x + threadIdx.x;
    if (i < n) dst[i] = (_Float16)src[i];
}

// ---------------------------------------------------------------------------
// Depthwise causal conv (k=4) + SiLU on xc -> u (f32 for scan, f16 for GEMM);
// SiLU(z) -> sz.  xz layout: [branch][MROWS][512] (branch 1 pre-flipped).
// ---------------------------------------------------------------------------
__device__ __forceinline__ float silu_f(float v) {
    return v * (1.0f / (1.0f + __expf(-v)));
}

__global__ void conv_silu_kernel(const float* __restrict__ xz,
                                 const float* __restrict__ f_cw, const float* __restrict__ f_cb,
                                 const float* __restrict__ b_cw, const float* __restrict__ b_cb,
                                 float* __restrict__ u, _Float16* __restrict__ uh,
                                 float* __restrict__ sz) {
    int t = blockIdx.x * blockDim.x + threadIdx.x;
    const int total = 2 * MROWS * D_INNER;
    if (t >= total) return;
    int d  = t % D_INNER;
    int m  = (t / D_INNER) % MROWS;
    int br = t / (D_INNER * MROWS);
    int l  = m % SEQ;

    const float* cw  = br ? b_cw : f_cw;
    const float* cb  = br ? b_cb : f_cb;
    const float* xzb = xz + (size_t)br * MROWS * (2 * D_INNER);

    float acc = cb[d];
#pragma unroll
    for (int j = 0; j < 4; ++j) {
        int ls = l - 3 + j;
        if (ls >= 0)
            acc += xzb[(size_t)(m - 3 + j) * (2 * D_INNER) + d] * cw[d * 4 + j];
    }
    size_t o  = (size_t)br * MROWS * D_INNER + (size_t)m * D_INNER + d;
    float  uv = silu_f(acc);
    u[o]  = uv;
    uh[o] = (_Float16)uv;
    float zv = xzb[(size_t)m * (2 * D_INNER) + D_INNER + d];
    sz[o] = silu_f(zv);
}

// ---------------------------------------------------------------------------
// Selective scan. One block per (branch, batch); thread d owns channel d with
// a 16-state recurrence in registers; per timestep the 40-wide dbc row
// (dt_r[8], B[16], C[16]) is broadcast through LDS.
// Emits gated = (y + u*D)*silu(z) directly as f16 (out-proj GEMM operand).
// ---------------------------------------------------------------------------
__global__ void scan_kernel(const float* __restrict__ dbc, const float* __restrict__ u,
                            const float* __restrict__ sz,
                            const float* __restrict__ f_dtw, const float* __restrict__ f_dtb,
                            const float* __restrict__ f_Al,  const float* __restrict__ f_Dp,
                            const float* __restrict__ b_dtw, const float* __restrict__ b_dtb,
                            const float* __restrict__ b_Al,  const float* __restrict__ b_Dp,
                            _Float16* __restrict__ gatedh) {
    int br = blockIdx.x >> 2;
    int bb = blockIdx.x & 3;
    int d  = threadIdx.x;                  // 0..255

    const float* dtw = br ? b_dtw : f_dtw;
    const float* dtb = br ? b_dtb : f_dtb;
    const float* Al  = br ? b_Al  : f_Al;
    const float* Dp  = br ? b_Dp  : f_Dp;

    float Areg[D_STATE], h[D_STATE], dtwr[DT_RANK];
#pragma unroll
    for (int s = 0; s < D_STATE; ++s) {
        Areg[s] = -__expf(Al[d * D_STATE + s]);
        h[s]    = 0.0f;
    }
#pragma unroll
    for (int r = 0; r < DT_RANK; ++r) dtwr[r] = dtw[d * DT_RANK + r];
    float dtbv = dtb[d];
    float Dd   = Dp[d];

    __shared__ float sh[DBC_LD];
    size_t rowbase = (size_t)br * MROWS + (size_t)bb * SEQ;

    for (int l = 0; l < SEQ; ++l) {
        size_t row = rowbase + l;
        if (threadIdx.x < (DT_RANK + 2 * D_STATE))
            sh[threadIdx.x] = dbc[row * DBC_LD + threadIdx.x];
        __syncthreads();

        float dtv = dtbv;
#pragma unroll
        for (int r = 0; r < DT_RANK; ++r) dtv += sh[r] * dtwr[r];
        dtv = (dtv > 20.0f) ? dtv : log1pf(__expf(dtv));   // softplus

        float x_t = u[row * D_INNER + d];
        float du  = dtv * x_t;
        float y   = 0.0f;
#pragma unroll
        for (int s = 0; s < D_STATE; ++s) {
            float dA = __expf(dtv * Areg[s]);
            h[s] = dA * h[s] + du * sh[DT_RANK + s];
            y += h[s] * sh[DT_RANK + D_STATE + s];
        }
        float yt = y + x_t * Dd;
        gatedh[row * D_INNER + d] = (_Float16)(yt * sz[row * D_INNER + d]);
        __syncthreads();
    }
}

// ---------------------------------------------------------------------------
// Wave32 LayerNorm helpers (one wave per 128-wide row, 4 elems/lane).
// ---------------------------------------------------------------------------
__device__ __forceinline__ float wave_sum(float v) {
#pragma unroll
    for (int off = 16; off > 0; off >>= 1) v += __shfl_xor(v, off, 32);
    return v;
}

__device__ __forceinline__ void ln_row(const float v[4], int lane,
                                       const float* __restrict__ g,
                                       const float* __restrict__ b, float o[4]) {
    float s = v[0] + v[1] + v[2] + v[3];
    float mu = wave_sum(s) * (1.0f / 128.0f);
    float q = 0.0f;
#pragma unroll
    for (int j = 0; j < 4; ++j) { float dd = v[j] - mu; q += dd * dd; }
    float inv = rsqrtf(wave_sum(q) * (1.0f / 128.0f) + 1e-5f);
#pragma unroll
    for (int j = 0; j < 4; ++j) {
        int i = lane + 32 * j;
        o[j] = (v[j] - mu) * inv * g[i] + b[i];
    }
}

// outs = LN(x + mo_f, for) + LN(x + flip(mo_b), back)   (f32 + f16 copies)
__global__ void dual_ln_kernel(const float* __restrict__ x, const float* __restrict__ mo,
                               const float* __restrict__ gF, const float* __restrict__ bF,
                               const float* __restrict__ gB, const float* __restrict__ bB,
                               float* __restrict__ outs, _Float16* __restrict__ outsh) {
    int wid  = (blockIdx.x * blockDim.x + threadIdx.x) >> 5;
    int lane = threadIdx.x & 31;
    if (wid >= MROWS) return;
    int b = wid >> 11;              // / SEQ
    int l = wid & (SEQ - 1);

    const float* xr = x  + (size_t)wid * D_MODEL;
    const float* mf = mo + (size_t)wid * D_MODEL;
    const float* mb = mo + (size_t)MROWS * D_MODEL
                         + (size_t)(b * SEQ + (SEQ - 1 - l)) * D_MODEL;

    float vf[4], vb[4], of[4], ob[4];
#pragma unroll
    for (int j = 0; j < 4; ++j) {
        int i = lane + 32 * j;
        float xv = xr[i];
        vf[j] = xv + mf[i];
        vb[j] = xv + mb[i];
    }
    ln_row(vf, lane, gF, bF, of);
    ln_row(vb, lane, gB, bB, ob);
#pragma unroll
    for (int j = 0; j < 4; ++j) {
        float ov = of[j] + ob[j];
        outs [(size_t)wid * D_MODEL + lane + 32 * j] = ov;
        outsh[(size_t)wid * D_MODEL + lane + 32 * j] = (_Float16)ov;
    }
}

// final = LN(outs + h2, ffn) -> d_out
__global__ void final_ln_kernel(const float* __restrict__ outs, const float* __restrict__ h2,
                                const float* __restrict__ g, const float* __restrict__ b,
                                float* __restrict__ out) {
    int wid  = (blockIdx.x * blockDim.x + threadIdx.x) >> 5;
    int lane = threadIdx.x & 31;
    if (wid >= MROWS) return;
    float v[4], o[4];
#pragma unroll
    for (int j = 0; j < 4; ++j) {
        int i = lane + 32 * j;
        v[j] = outs[(size_t)wid * D_MODEL + i] + h2[(size_t)wid * D_MODEL + i];
    }
    ln_row(v, lane, g, b, o);
#pragma unroll
    for (int j = 0; j < 4; ++j)
        out[(size_t)wid * D_MODEL + lane + 32 * j] = o[j];
}

// ---------------------------------------------------------------------------
// Host-side orchestration
// ---------------------------------------------------------------------------
static inline int gemm_blocks(int Mr, int Nr, int ntile) {
    int ngrp  = (Nr + 16 * ntile - 1) / (16 * ntile);
    int waves = ((Mr + 15) / 16) * ngrp;
    return (waves * 32 + 255) / 256;
}

static inline void launch_cvt(const float* s, _Float16* d, int n, hipStream_t st) {
    cvt_f16_kernel<<<(n + 255) / 256, 256, 0, st>>>(s, d, n);
}

extern "C" void kernel_launch(void* const* d_in, const int* in_sizes, int n_in,
                              void* d_out, int out_size, void* d_ws, size_t ws_size,
                              hipStream_t stream) {
    const float* x         = (const float*)d_in[0];
    const float* mf_in_w   = (const float*)d_in[1];
    const float* mf_conv_w = (const float*)d_in[2];
    const float* mf_conv_b = (const float*)d_in[3];
    const float* mf_xproj  = (const float*)d_in[4];
    const float* mf_dt_w   = (const float*)d_in[5];
    const float* mf_dt_b   = (const float*)d_in[6];
    const float* mf_A_log  = (const float*)d_in[7];
    const float* mf_D      = (const float*)d_in[8];
    const float* mf_out_w  = (const float*)d_in[9];
    const float* mb_in_w   = (const float*)d_in[10];
    const float* mb_conv_w = (const float*)d_in[11];
    const float* mb_conv_b = (const float*)d_in[12];
    const float* mb_xproj  = (const float*)d_in[13];
    const float* mb_dt_w   = (const float*)d_in[14];
    const float* mb_dt_b   = (const float*)d_in[15];
    const float* mb_A_log  = (const float*)d_in[16];
    const float* mb_D      = (const float*)d_in[17];
    const float* mb_out_w  = (const float*)d_in[18];
    const float* ln_for_g  = (const float*)d_in[19];
    const float* ln_for_b  = (const float*)d_in[20];
    const float* ln_back_g = (const float*)d_in[21];
    const float* ln_back_b = (const float*)d_in[22];
    const float* ln_ffn_g  = (const float*)d_in[23];
    const float* ln_ffn_b  = (const float*)d_in[24];
    const float* ffn_w1    = (const float*)d_in[25];
    const float* ffn_b1    = (const float*)d_in[26];
    const float* ffn_w2    = (const float*)d_in[27];
    const float* ffn_b2    = (const float*)d_in[28];

    const int M = MROWS;

    // f32 scratch region
    float* ws    = (float*)d_ws;
    float* xz    = ws;                                   // [2][M][512]
    float* u     = xz    + (size_t)2 * M * 512;          // [2][M][256]
    float* sz    = u     + (size_t)2 * M * 256;          // [2][M][256]
    float* dbc   = sz    + (size_t)2 * M * 256;          // [2][M][48]
    float* mo    = dbc   + (size_t)2 * M * DBC_LD;       // [2][M][128]
    float* outs  = mo    + (size_t)2 * M * 128;          // [M][128]
    float* h2    = outs  + (size_t)M * 128;              // [M][128]
    float* f32end= h2    + (size_t)M * 128;

    // f16 scratch region
    _Float16* hp      = (_Float16*)f32end;
    _Float16* xh      = hp;                              // [M][128]
    _Float16* uh      = xh      + (size_t)M * 128;       // [2][M][256]
    _Float16* gatedh  = uh      + (size_t)2 * M * 256;   // [2][M][256]
    _Float16* outsh   = gatedh  + (size_t)2 * M * 256;   // [M][128]
    _Float16* h1h     = outsh   + (size_t)M * 128;       // [M][256]
    _Float16* inwh_f  = h1h     + (size_t)M * 256;       // [512][128]
    _Float16* inwh_b  = inwh_f  + 512 * 128;
    _Float16* xprojh_f= inwh_b  + 512 * 128;             // [40][256]
    _Float16* xprojh_b= xprojh_f+ 40 * 256;
    _Float16* outwh_f = xprojh_b+ 40 * 256;              // [128][256]
    _Float16* outwh_b = outwh_f + 128 * 256;
    _Float16* ffn1h   = outwh_b + 128 * 256;             // [256][128]
    _Float16* ffn2h   = ffn1h   + 256 * 128;             // [128][256]

    // 0) one-shot f32 -> f16 conversions (x + weights)
    launch_cvt(x,        xh,       M * 128,   stream);
    launch_cvt(mf_in_w,  inwh_f,   512 * 128, stream);
    launch_cvt(mb_in_w,  inwh_b,   512 * 128, stream);
    launch_cvt(mf_xproj, xprojh_f, 40 * 256,  stream);
    launch_cvt(mb_xproj, xprojh_b, 40 * 256,  stream);
    launch_cvt(mf_out_w, outwh_f,  128 * 256, stream);
    launch_cvt(mb_out_w, outwh_b,  128 * 256, stream);
    launch_cvt(ffn_w1,   ffn1h,    256 * 128, stream);
    launch_cvt(ffn_w2,   ffn2h,    128 * 256, stream);

    // 1) in-projection GEMMs (backward branch reads flipped rows)
    wmma_gemm<0, 4, 0><<<gemm_blocks(M, 512, 4), 256, 0, stream>>>(
        xh, inwh_f, nullptr, xz, nullptr, M, 512, 128, 128, 128, 512, 0);
    wmma_gemm<0, 4, 0><<<gemm_blocks(M, 512, 4), 256, 0, stream>>>(
        xh, inwh_b, nullptr, xz + (size_t)M * 512, nullptr, M, 512, 128, 128, 128, 512, SEQ);

    // 2) depthwise conv + SiLU -> u (f32+f16), SiLU(z) -> sz
    {
        int total  = 2 * M * D_INNER;
        conv_silu_kernel<<<(total + 255) / 256, 256, 0, stream>>>(
            xz, mf_conv_w, mf_conv_b, mb_conv_w, mb_conv_b, u, uh, sz);
    }

    // 3) x-projection GEMMs: dbc = u @ xproj^T  (N=40, 3 N-tiles, clamped cols)
    wmma_gemm<0, 3, 0><<<gemm_blocks(M, 40, 3), 256, 0, stream>>>(
        uh, xprojh_f, nullptr, dbc, nullptr, M, 40, 256, 256, 256, DBC_LD, 0);
    wmma_gemm<0, 3, 0><<<gemm_blocks(M, 40, 3), 256, 0, stream>>>(
        uh + (size_t)M * 256, xprojh_b, nullptr, dbc + (size_t)M * DBC_LD, nullptr,
        M, 40, 256, 256, 256, DBC_LD, 0);

    // 4) selective scan (dt projection + softplus fused) -> gated (f16)
    scan_kernel<<<8, 256, 0, stream>>>(dbc, u, sz,
                                       mf_dt_w, mf_dt_b, mf_A_log, mf_D,
                                       mb_dt_w, mb_dt_b, mb_A_log, mb_D, gatedh);

    // 5) out-projection GEMMs -> mo (f32)
    wmma_gemm<0, 4, 0><<<gemm_blocks(M, 128, 4), 256, 0, stream>>>(
        gatedh, outwh_f, nullptr, mo, nullptr, M, 128, 256, 256, 256, 128, 0);
    wmma_gemm<0, 4, 0><<<gemm_blocks(M, 128, 4), 256, 0, stream>>>(
        gatedh + (size_t)M * 256, outwh_b, nullptr, mo + (size_t)M * 128, nullptr,
        M, 128, 256, 256, 256, 128, 0);

    // 6) dual residual LayerNorm (flip-back inside) -> outs (f32+f16)
    dual_ln_kernel<<<(M * 32 + 255) / 256, 256, 0, stream>>>(
        x, mo, ln_for_g, ln_for_b, ln_back_g, ln_back_b, outs, outsh);

    // 7) FFN: h1 = relu(outs @ w1^T + b1) (f16 only); h2 = h1 @ w2^T + b2 (f32)
    wmma_gemm<1, 4, 1><<<gemm_blocks(M, 256, 4), 256, 0, stream>>>(
        outsh, ffn1h, ffn_b1, nullptr, h1h, M, 256, 128, 128, 128, 256, 0);
    wmma_gemm<2, 4, 0><<<gemm_blocks(M, 128, 4), 256, 0, stream>>>(
        h1h, ffn2h, ffn_b2, h2, nullptr, M, 128, 256, 256, 256, 128, 0);

    // 8) final LayerNorm -> d_out
    final_ln_kernel<<<(M * 32 + 255) / 256, 256, 0, stream>>>(
        outs, h2, ln_ffn_g, ln_ffn_b, (float*)d_out);
}
